// MHABlock_87574383165700
// MI455X (gfx1250) — matrix operand order repaired and checked by reading, not compiled, not run on hardware
//
#include <hip/hip_runtime.h>
#include <math.h>

typedef __bf16 bf16;
typedef __attribute__((ext_vector_type(16))) __bf16 v16bf;
typedef __attribute__((ext_vector_type(8)))  float  v8f;

#define CC   128      // channels
#define BB   8        // batch
#define TS   8        // spatial positions per workgroup
#define MT   64       // tokens per workgroup = BB*TS
#define HWN  16384    // 128*128 spatial
#define NT   256      // threads per block

// padded LDS strides (row-to-row = 272B / 528B -> bank step 4 dwords, no conflicts)
#define AST  136      // bf16 activation stride
#define WST  136      // bf16 weight stride
#define HST  264      // bf16 double-width (HID) stride
#define TST  132      // f32 residual stride

#define NCHUNK 15     // packed 128x128 bf16 weight chunks in d_ws arena

// ---------------- LDS layout (dynamic, ~162.8 KB) ----------------
struct SM {
  float T [MT*TST];       // f32 residual "t"
  bf16  A [MT*AST];       // GEMM input (ln outputs / normalized x)
  bf16  WB[CC*WST];       // staged weight (bf16, row-major [N][K] chunk)
  bf16  P0[MT*AST];       // q2 / bf16(t)
  bf16  P1[MT*AST];       // chained temp / attention z
  bf16  Hb[MT*HST];       // k2|v2, then a|gelu-product
  float gnm[64], gnr[64];
  float gng[CC], gnbv[CC];
  float cibv[CC], pe8[BB*CC];
  float l1g[CC], l1b[CC], l2g[CC], l2b[CC];
};

// ---------------- WMMA fragment helpers (contiguous k order, K = k0 + 16*half + e, in both A and B) ----------------
__device__ __forceinline__ v8f vzero() {
  v8f z;
#pragma unroll
  for (int i = 0; i < 8; ++i) z[i] = 0.f;
  return z;
}

// A-matrix 16x32 bf16: lane L -> row M = m0 + (L&15), K-half = L>>4.
// element e: K = k0 + 16*kh + e
__device__ __forceinline__ v16bf load_a_frag(const bf16* __restrict__ base, int stride,
                                             int m0, int k0, int lane) {
  const int row = m0 + (lane & 15);
  const int kh  = (lane >> 4) & 1;
  const bf16* p = base + row * stride + k0 + kh * 16;
  v16bf a;
#pragma unroll
  for (int e = 0; e < 8; ++e) { a[e] = p[e]; a[e + 8] = p[e + 8]; }
  return a;
}

// B-matrix 32x16 bf16 from weight W[n][k] (B(k,n)=W[n][k]):
// lane L -> col N = n0 + (L&15), K-half = L>>4; element e: K = k0 + 16*kh + e
__device__ __forceinline__ v16bf load_b_frag(const bf16* __restrict__ w, int n0, int k0, int lane) {
  const int nn = n0 + (lane & 15);
  const int kh = (lane >> 4) & 1;
  const bf16* p = w + nn * WST + k0 + kh * 16;
  v16bf b;
#pragma unroll
  for (int e = 0; e < 16; ++e) b[e] = p[e];
  return b;
}

// One 16x16 output tile, K=128 (4 WMMA steps). ak0 = K offset in activation buf.
__device__ __forceinline__ void gemm_k(v8f& acc, const bf16* __restrict__ Ab, int astr, int ak0,
                                       const bf16* __restrict__ Wb, int m0, int n0, int lane) {
#pragma unroll
  for (int kk = 0; kk < 4; ++kk) {
    v16bf a = load_a_frag(Ab, astr, m0, ak0 + kk * 32, lane);
    v16bf b = load_b_frag(Wb, n0, kk * 32, lane);
    acc = __builtin_amdgcn_wmma_f32_16x16x32_bf16(false, a, false, b, (short)0, acc, false, false);
  }
}

// Stage a pre-packed 128x128 bf16 chunk from the global arena into WB (16B copies)
__device__ __forceinline__ void stage_chunk(bf16* __restrict__ dst, const bf16* __restrict__ arena,
                                            int chunk, int tid) {
  const uint4* s = (const uint4*)(arena + (size_t)chunk * (CC * CC));
  for (int idx = tid; idx < CC * CC / 8; idx += NT) {   // 2048 x uint4
    int r = idx >> 4, c8 = idx & 15;
    *(uint4*)&dst[r * WST + c8 * 8] = s[idx];
  }
}

__device__ __forceinline__ float gelu_exact(float v) {
  return 0.5f * v * (1.0f + erff(v * 0.70710678118654752440f));
}

// stage chunk; GEMM src(bf16,[MT][sstr]) x W^T + bias(global) -> dst bf16 at column dcol0
__device__ __forceinline__ void std_gemm(int chunk, const bf16* arena,
                                         const bf16* src, int sstr,
                                         bf16* dst, int dstr, int dcol0,
                                         const float* __restrict__ bias_g, int bias0,
                                         SM& sm, int tid, int lane, int m0, int nb) {
  __syncthreads();
  stage_chunk(sm.WB, arena, chunk, tid);
  __syncthreads();
  const int lcol = lane & 15;
  const int rb   = ((lane >> 4) & 1) * 8;
  for (int nt = 0; nt < 4; ++nt) {
    int n0 = (nb + nt) * 16;
    v8f acc = vzero();
    gemm_k(acc, src, sstr, 0, sm.WB, m0, n0, lane);
    int col = n0 + lcol;
    float bv = bias_g[bias0 + col];
#pragma unroll
    for (int r = 0; r < 8; ++r)
      dst[(m0 + rb + r) * dstr + dcol0 + col] = (bf16)(acc[r] + bv);
  }
}

// ---------------- Kernel W: pack all weights into bf16 arena (15 chunks) ----------------
__global__ __launch_bounds__(NT) void pack_weights_kernel(
    const float* __restrict__ ciw, const float* __restrict__ wq, const float* __restrict__ win,
    const float* __restrict__ wk, const float* __restrict__ wv, const float* __restrict__ wout,
    const float* __restrict__ w1, const float* __restrict__ w2, const float* __restrict__ w3,
    const float* __restrict__ cow, bf16* __restrict__ arena) {
  const float* src; int row0 = 0, rowlen = CC, k0 = 0;
  switch (blockIdx.x) {
    case 0:  src = ciw;  break;
    case 1:  src = wq;   break;
    case 2:  src = win;  break;                    // wiq
    case 3:  src = wk;   break;
    case 4:  src = win;  row0 = 128; break;        // wik
    case 5:  src = wv;   break;
    case 6:  src = win;  row0 = 256; break;        // wiv
    case 7:  src = wout; break;
    case 8:  src = w1;   break;
    case 9:  src = w1;   row0 = 128; break;
    case 10: src = w2;   break;
    case 11: src = w2;   row0 = 128; break;
    case 12: src = w3;   rowlen = 256; break;      // w3 K-chunk 0
    case 13: src = w3;   rowlen = 256; k0 = 128; break;
    default: src = cow;  break;
  }
  bf16* dst = arena + (size_t)blockIdx.x * (CC * CC);
  for (int idx = threadIdx.x; idx < CC * CC; idx += NT) {
    int r = idx >> 7, c = idx & 127;
    dst[idx] = (bf16)src[(size_t)(row0 + r) * rowlen + k0 + c];
  }
}

// ---------------- Kernel A: GroupNorm statistics ----------------
__global__ __launch_bounds__(NT) void gn_stats_kernel(const float* __restrict__ x,
                                                      float* __restrict__ ws) {
  __shared__ float ssum[NT], ssq[NT];
  const int bid = blockIdx.x;                 // (b*8 + g), 64 blocks
  const float* p = x + (size_t)bid * (16 * HWN);
  float s = 0.f, q = 0.f;
  for (int i = threadIdx.x; i < 16 * HWN; i += NT) { float v = p[i]; s += v; q += v * v; }
  ssum[threadIdx.x] = s; ssq[threadIdx.x] = q;
  __syncthreads();
  for (int st = NT / 2; st > 0; st >>= 1) {
    if (threadIdx.x < st) { ssum[threadIdx.x] += ssum[threadIdx.x + st];
                            ssq[threadIdx.x]  += ssq[threadIdx.x + st]; }
    __syncthreads();
  }
  if (threadIdx.x == 0) {
    const float invn = 1.f / (16.f * HWN);
    float mean = ssum[0] * invn;
    float var  = ssq[0] * invn - mean * mean;
    ws[bid] = mean;
    ws[64 + bid] = rsqrtf(var + 1e-6f);
  }
}

// ---------------- Kernel B: fully fused transformer block ----------------
__global__ __launch_bounds__(NT) void fused_block_kernel(
    const float* __restrict__ x,  const float* __restrict__ gn_g, const float* __restrict__ gn_b,
    const float* __restrict__ cib, const float* __restrict__ pe,
    const float* __restrict__ ln1g, const float* __restrict__ ln1b,
    const float* __restrict__ bq, const float* __restrict__ bk, const float* __restrict__ bv,
    const float* __restrict__ bin_, const float* __restrict__ bout,
    const float* __restrict__ ln2g, const float* __restrict__ ln2b,
    const float* __restrict__ b1, const float* __restrict__ b2, const float* __restrict__ b3,
    const float* __restrict__ cob,
    const float* __restrict__ ws, const bf16* __restrict__ arena, float* __restrict__ out) {
  extern __shared__ char smem_raw[];
  SM& sm = *reinterpret_cast<SM*>(smem_raw);

  const int tid  = threadIdx.x;
  const int lane = tid & 31;
  const int wid  = tid >> 5;
  const int m0   = (wid >> 1) * 16;       // 4 M-tiles, 2 waves each
  const int nb   = (wid & 1) * 4;         // 4 N-tiles per wave
  const int lcol = lane & 15;
  const int rb   = ((lane >> 4) & 1) * 8;
  const int n0s  = blockIdx.x * TS;       // spatial base

  // ---- stage small params ----
  for (int i = tid; i < 64; i += NT) { sm.gnm[i] = ws[i]; sm.gnr[i] = ws[64 + i]; }
  for (int i = tid; i < CC; i += NT) {
    sm.gng[i] = gn_g[i]; sm.gnbv[i] = gn_b[i]; sm.cibv[i] = cib[i];
    sm.l1g[i] = ln1g[i]; sm.l1b[i] = ln1b[i]; sm.l2g[i] = ln2g[i]; sm.l2b[i] = ln2b[i];
  }
  for (int i = tid; i < BB * CC; i += NT) sm.pe8[i] = pe[i];
  // prefetch first weight chunks into GL2 (global_prefetch_b8)
  __builtin_prefetch(arena, 0, 1);
  __builtin_prefetch(arena + 2 * CC * CC, 0, 1);
  __builtin_prefetch(arena + 4 * CC * CC, 0, 1);
  __syncthreads();

  // ---- Phase 1: load x tile, GroupNorm-normalize -> A (bf16) ----
  for (int idx = tid; idx < BB * CC * TS; idx += NT) {
    int s = idx & 7, c = (idx >> 3) & 127, b = idx >> 10;
    float v = x[(((size_t)(b * CC + c)) << 14) + n0s + s];
    int g = c >> 4;
    float xn = (v - sm.gnm[b * 8 + g]) * sm.gnr[b * 8 + g] * sm.gng[c] + sm.gnbv[c];
    sm.A[(b * TS + s) * AST + c] = (bf16)xn;
  }

  // ---- Phase 2: conv_in: t = Xn*ciw^T + cib + pe[b]  -> T (f32) ----
  __syncthreads();
  stage_chunk(sm.WB, arena, 0, tid);
  __syncthreads();
  {
    const int bl = (m0 + rb) >> 3;   // batch index, lane-uniform
    for (int nt = 0; nt < 4; ++nt) {
      int n0 = (nb + nt) * 16;
      v8f acc = vzero();
      gemm_k(acc, sm.A, AST, 0, sm.WB, m0, n0, lane);
      int col = n0 + lcol;
      float add = sm.cibv[col] + sm.pe8[bl * CC + col];
#pragma unroll
      for (int r = 0; r < 8; ++r) sm.T[(m0 + rb + r) * TST + col] = acc[r] + add;
    }
  }
  __syncthreads();

  // ---- Phase 3: ln1(t) -> A ----
  if (tid < MT) {
    const float* row = &sm.T[tid * TST];
    float mu = 0.f;
    for (int c = 0; c < CC; ++c) mu += row[c];
    mu *= (1.f / CC);
    float var = 0.f;
    for (int c = 0; c < CC; ++c) { float d = row[c] - mu; var += d * d; }
    float rs = rsqrtf(var * (1.f / CC) + 1e-5f);
    for (int c = 0; c < CC; ++c)
      sm.A[tid * AST + c] = (bf16)((row[c] - mu) * rs * sm.l1g[c] + sm.l1b[c]);
  }

  // ---- Phases 4-5: chained q/k/v projections ----
  std_gemm(1, arena, sm.A,  AST, sm.P1, AST, 0,   bq,   0,   sm, tid, lane, m0, nb);
  std_gemm(2, arena, sm.P1, AST, sm.P0, AST, 0,   bin_, 0,   sm, tid, lane, m0, nb); // q2
  std_gemm(3, arena, sm.A,  AST, sm.P1, AST, 0,   bk,   0,   sm, tid, lane, m0, nb);
  std_gemm(4, arena, sm.P1, AST, sm.Hb, HST, 0,   bin_, CC,  sm, tid, lane, m0, nb); // k2
  std_gemm(5, arena, sm.A,  AST, sm.P1, AST, 0,   bv,   0,   sm, tid, lane, m0, nb);
  std_gemm(6, arena, sm.P1, AST, sm.Hb, HST, CC,  bin_, 2*CC, sm, tid, lane, m0, nb); // v2
  __syncthreads();

  // ---- Phase 6: attention over the 8-batch "sequence" (seq len 8, hd 16) ----
  if (tid < MT) {
    const int sid = tid >> 3, h = tid & 7, ch = h * 16;
    float qi[16], sc[8];
    for (int i = 0; i < BB; ++i) {
      for (int d = 0; d < 16; ++d) qi[d] = (float)sm.P0[(i * TS + sid) * AST + ch + d];
      float mx = -3.0e38f;
      for (int j = 0; j < BB; ++j) {
        float s = 0.f;
        for (int d = 0; d < 16; ++d)
          s += qi[d] * (float)sm.Hb[(j * TS + sid) * HST + ch + d];
        s *= 0.25f;                 // 1/sqrt(16)
        sc[j] = s; mx = fmaxf(mx, s);
      }
      float den = 0.f;
      for (int j = 0; j < BB; ++j) { sc[j] = expf(sc[j] - mx); den += sc[j]; }
      float inv = 1.f / den;
      for (int d = 0; d < 16; ++d) {
        float z = 0.f;
        for (int j = 0; j < BB; ++j)
          z += sc[j] * (float)sm.Hb[(j * TS + sid) * HST + CC + ch + d];
        sm.P1[(i * TS + sid) * AST + ch + d] = (bf16)(z * inv);
      }
    }
  }

  // ---- Phase 7: wout + residual into T ----
  __syncthreads();
  stage_chunk(sm.WB, arena, 7, tid);
  __syncthreads();
  for (int nt = 0; nt < 4; ++nt) {
    int n0 = (nb + nt) * 16;
    v8f acc = vzero();
    gemm_k(acc, sm.P1, AST, 0, sm.WB, m0, n0, lane);
    int col = n0 + lcol;
    float bv = bout[col];
#pragma unroll
    for (int r = 0; r < 8; ++r)
      sm.T[(m0 + rb + r) * TST + col] += acc[r] + bv;
  }
  __syncthreads();

  // ---- Phase 8: ln2(t) -> A ; bf16(t) -> P0 (gate input) ----
  if (tid < MT) {
    const float* row = &sm.T[tid * TST];
    float mu = 0.f;
    for (int c = 0; c < CC; ++c) mu += row[c];
    mu *= (1.f / CC);
    float var = 0.f;
    for (int c = 0; c < CC; ++c) { float d = row[c] - mu; var += d * d; }
    float rs = rsqrtf(var * (1.f / CC) + 1e-5f);
    for (int c = 0; c < CC; ++c) {
      float v = row[c];
      sm.A[tid * AST + c]  = (bf16)((v - mu) * rs * sm.l2g[c] + sm.l2b[c]);
      sm.P0[tid * AST + c] = (bf16)v;
    }
  }

  // ---- Phase 9: a = tn2*w1^T + b1 (HID=256, two halves) -> Hb ----
  std_gemm(8, arena, sm.A, AST, sm.Hb, HST, 0,  b1, 0,  sm, tid, lane, m0, nb);
  std_gemm(9, arena, sm.A, AST, sm.Hb, HST, CC, b1, CC, sm, tid, lane, m0, nb);

  // ---- Phase 10: g = t*w2^T + b2 ; Hb <- a * gelu(g) ----
  for (int half = 0; half < 2; ++half) {
    __syncthreads();
    stage_chunk(sm.WB, arena, 10 + half, tid);
    __syncthreads();
    for (int nt = 0; nt < 4; ++nt) {
      int n0 = (nb + nt) * 16;
      v8f acc = vzero();
      gemm_k(acc, sm.P0, AST, 0, sm.WB, m0, n0, lane);
      int col = n0 + lcol;
      float bc = b2[half * CC + col];
#pragma unroll
      for (int r = 0; r < 8; ++r) {
        int off = (m0 + rb + r) * HST + half * CC + col;
        float gv = acc[r] + bc;
        float av = (float)sm.Hb[off];
        sm.Hb[off] = (bf16)(av * gelu_exact(gv));
      }
    }
  }

  // ---- Phase 11: m = h*w3^T + b3 (K=256, staged in two K-chunks); t += m ----
  {
    v8f accs[4];
#pragma unroll
    for (int nt = 0; nt < 4; ++nt) accs[nt] = vzero();
    __syncthreads();
    stage_chunk(sm.WB, arena, 12, tid);
    __syncthreads();
    for (int nt = 0; nt < 4; ++nt)
      gemm_k(accs[nt], sm.Hb, HST, 0, sm.WB, m0, (nb + nt) * 16, lane);
    __syncthreads();
    stage_chunk(sm.WB, arena, 13, tid);
    __syncthreads();
    for (int nt = 0; nt < 4; ++nt)
      gemm_k(accs[nt], sm.Hb, HST, CC, sm.WB, m0, (nb + nt) * 16, lane);
    for (int nt = 0; nt < 4; ++nt) {
      int col = (nb + nt) * 16 + lcol;
      float bc = b3[col];
#pragma unroll
      for (int r = 0; r < 8; ++r)
        sm.T[(m0 + rb + r) * TST + col] += accs[nt][r] + bc;
    }
  }
  __syncthreads();

  // ---- Phase 12: conv_out + cob, then add long residual x and store ----
  for (int idx = tid; idx < MT * CC; idx += NT) {
    int t = idx >> 7, c = idx & 127;
    sm.P0[t * AST + c] = (bf16)sm.T[t * TST + c];
  }
  __syncthreads();
  stage_chunk(sm.WB, arena, 14, tid);
  __syncthreads();
  for (int nt = 0; nt < 4; ++nt) {
    int n0 = (nb + nt) * 16;
    v8f acc = vzero();
    gemm_k(acc, sm.P0, AST, 0, sm.WB, m0, n0, lane);
    int col = n0 + lcol;
    float bv = cob[col];
#pragma unroll
    for (int r = 0; r < 8; ++r)
      sm.T[(m0 + rb + r) * TST + col] = acc[r] + bv;
  }
  __syncthreads();
  for (int idx = tid; idx < BB * CC * TS; idx += NT) {
    int s = idx & 7, c = (idx >> 3) & 127, b = idx >> 10;
    size_t gi = (((size_t)(b * CC + c)) << 14) + n0s + s;
    out[gi] = sm.T[(b * TS + s) * TST + c] + x[gi];
  }
}

// ---------------- Host launcher ----------------
extern "C" void kernel_launch(void* const* d_in, const int* in_sizes, int n_in,
                              void* d_out, int out_size, void* d_ws, size_t ws_size,
                              hipStream_t stream) {
  (void)in_sizes; (void)n_in; (void)out_size; (void)ws_size;
  const float* x    = (const float*)d_in[0];
  const float* gn_g = (const float*)d_in[1];
  const float* gn_b = (const float*)d_in[2];
  const float* ciw  = (const float*)d_in[3];
  const float* cib  = (const float*)d_in[4];
  const float* pe   = (const float*)d_in[5];
  const float* ln1g = (const float*)d_in[6];
  const float* ln1b = (const float*)d_in[7];
  const float* wq   = (const float*)d_in[8];
  const float* bq   = (const float*)d_in[9];
  const float* wk   = (const float*)d_in[10];
  const float* bk   = (const float*)d_in[11];
  const float* wv   = (const float*)d_in[12];
  const float* bv   = (const float*)d_in[13];
  const float* win  = (const float*)d_in[14];
  const float* bin_ = (const float*)d_in[15];
  const float* wout = (const float*)d_in[16];
  const float* bout = (const float*)d_in[17];
  const float* ln2g = (const float*)d_in[18];
  const float* ln2b = (const float*)d_in[19];
  const float* w1   = (const float*)d_in[20];
  const float* b1   = (const float*)d_in[21];
  const float* w2   = (const float*)d_in[22];
  const float* b2   = (const float*)d_in[23];
  const float* w3   = (const float*)d_in[24];
  const float* b3   = (const float*)d_in[25];
  const float* cow  = (const float*)d_in[26];
  const float* cob  = (const float*)d_in[27];

  float* ws   = (float*)d_ws;                       // [0..127]: GN stats
  bf16* arena = (bf16*)((char*)d_ws + 512);         // 15 x 128x128 bf16 chunks (~480 KB)
  float* out  = (float*)d_out;

  pack_weights_kernel<<<NCHUNK, NT, 0, stream>>>(ciw, wq, win, wk, wv, wout, w1, w2, w3, cow, arena);
  gn_stats_kernel<<<64, NT, 0, stream>>>(x, ws);
  fused_block_kernel<<<HWN / TS, NT, sizeof(SM), stream>>>(
      x, gn_g, gn_b, cib, pe, ln1g, ln1b, bq, bk, bv, bin_, bout,
      ln2g, ln2b, b1, b2, b3, cob, ws, arena, out);
}
